// flow_mapping_73461120631363
// MI455X (gfx1250) — matrix operand order repaired and checked by reading, not compiled, not run on hardware
//
#include <hip/hip_runtime.h>
#include <hip/hip_bf16.h>
#include <math.h>

// Problem constants (from reference)
#define BB 65536
#define LL 64
#define NS 32
#define NW 256
#define DD 8
#define ALPHA 0.6f

typedef __attribute__((ext_vector_type(16))) __bf16 v16bf;
typedef __attribute__((ext_vector_type(8)))  __bf16 v8bf;
typedef __attribute__((ext_vector_type(8)))  float  v8f;

// -------- LDS layout (dynamic shared) --------
// A1 : 64 x 32 bf16, row stride 40  (80 B)          5120 B   @ 0
// H1 : 64 x 256 bf16, row stride 264 (528 B)       33792 B   @ 5120    (H3 f32 aliases)
// H2 : 64 x 256 bf16, row stride 264 (528 B)       33792 B   @ 38912
// Z0 : 64 x 64 f32, row stride 68 (272 B)          17408 B   @ 72704
// Z1 : same                                        17408 B   @ 90112
#define A1_OFF   0
#define H1_OFF   5120
#define H2_OFF   38912
#define Z0_OFF   72704
#define Z1_OFF   90112
#define SMEM_BYTES (90112 + 17408)
#define SA  40
#define SH  264
#define S3  72
#define SZ  68

// ---------------- weight pre-pack ----------------
// dst layout: [layer][ntile][ktile][lane(32)][16 bf16]
// lane: n = nt*16 + (lane&15);  K = kt*32 + (lane>>4)*8 + e + (e>>3)*8
__global__ void pack_w_kernel(const float* __restrict__ W, __bf16* __restrict__ dst,
                              int Ktot, int N, int total) {
    int idx = blockIdx.x * blockDim.x + threadIdx.x;
    if (idx >= total) return;
    int lane = idx & 31;
    int tile = idx >> 5;
    int ktiles = Ktot >> 5;
    int ntiles = N >> 4;
    int kt = tile % ktiles; tile /= ktiles;
    int nt = tile % ntiles; tile /= ntiles;
    int l  = tile;
    int n  = nt * 16 + (lane & 15);
    int kb = kt * 32 + ((lane >> 4) << 3);
    __bf16* o = dst + ((((size_t)(l * ntiles + nt) * ktiles + kt) * 32 + lane) * 16);
    #pragma unroll
    for (int e = 0; e < 16; ++e)
        o[e] = (__bf16)W[((size_t)l * Ktot + (kb + e + ((e >> 3) << 3))) * N + n];
}

// per-layer sum(logs) -> ws scalar table
__global__ void sumlogs_kernel(const float* __restrict__ logs, float* __restrict__ o) {
    int l = threadIdx.x;
    if (l < DD) {
        float s = 0.f;
        for (int j = 0; j < LL; ++j) s += logs[l * LL + j];
        o[l] = s;
    }
}

// ---------------- fragment loaders ----------------
// 16-bit A 16x32 layout: lanes 0-15 -> M=lane, K {kb..kb+7} in v0-3, {kb+16..kb+23} in v4-7;
// lanes 16-31 -> +8 K offset.  `p` must already point at row (lane&15), K-half offset.
__device__ __forceinline__ v16bf load_a_at(const __bf16* p) {
    union { v16bf v; v8bf h[2]; } u;
    u.h[0] = *(const v8bf*)p;
    u.h[1] = *(const v8bf*)(p + 16);
    return u.v;
}

__device__ __forceinline__ v16bf load_b_at(const __bf16* p) {
    union { v16bf v; v8bf h[2]; } u;
    u.h[0] = *(const v8bf*)p;
    u.h[1] = *(const v8bf*)(p + 8);
    return u.v;
}

// ---------------- GEMM stages ----------------
// C/D layout: VGPR e -> row mbase + e + ((lane>>4)<<3), col = ntile*16 + (lane&15)
// GEMM1/2: each wave owns two adjacent N-tiles (shared A frag, two wmma chains).
template<int KTILES>
__device__ __forceinline__ void gemm_nt2(const __bf16* __restrict__ Abuf, int astride,
                                         const __bf16* __restrict__ Bpack,
                                         const float* __restrict__ bias,
                                         __bf16* __restrict__ Obuf,
                                         int wave, int lane) {
    const int nt0 = wave * 2, nt1 = nt0 + 1;
    const int nc  = lane & 15;
    const float bv0 = bias[nt0 * 16 + nc];
    const float bv1 = bias[nt1 * 16 + nc];
    // hoisted per-lane bases
    const __bf16* b0base = Bpack + (size_t)nt0 * KTILES * 512 + lane * 16;
    const __bf16* b1base = Bpack + (size_t)nt1 * KTILES * 512 + lane * 16;
    const __bf16* abase  = Abuf + (lane & 15) * astride + ((lane >> 4) << 3);
    __bf16* o0base = Obuf + ((lane >> 4) << 3) * SH + nt0 * 16 + nc;
    __bf16* o1base = Obuf + ((lane >> 4) << 3) * SH + nt1 * 16 + nc;

    #pragma unroll 1                  // real loop: keeps live-range small, no spills
    for (int mt = 0; mt < 4; ++mt) {
        const __bf16* arow = abase + mt * 16 * astride;
        v8f acc0, acc1;
        #pragma unroll
        for (int e = 0; e < 8; ++e) { acc0[e] = bv0; acc1[e] = bv1; }
        #pragma unroll
        for (int kt = 0; kt < KTILES; ++kt) {
            v16bf a  = load_a_at(arow + kt * 32);
            v16bf b0 = load_b_at(b0base + kt * 512);
            v16bf b1 = load_b_at(b1base + kt * 512);
            acc0 = __builtin_amdgcn_wmma_f32_16x16x32_bf16(false, a, false, b0,
                                                           (short)0, acc0, false, false);
            acc1 = __builtin_amdgcn_wmma_f32_16x16x32_bf16(false, a, false, b1,
                                                           (short)0, acc1, false, false);
        }
        __bf16* o0 = o0base + mt * 16 * SH;
        __bf16* o1 = o1base + mt * 16 * SH;
        #pragma unroll
        for (int e = 0; e < 8; ++e) {
            o0[e * SH] = (__bf16)fmaxf(acc0[e], 0.f);
            o1[e * SH] = (__bf16)fmaxf(acc1[e], 0.f);
        }
    }
}

// GEMM3: each wave owns one N-tile and two M-tiles (shared B frag), f32 out.
__device__ __forceinline__ void gemm3_stage(const __bf16* __restrict__ Abuf,
                                            const __bf16* __restrict__ Bpack,
                                            const float* __restrict__ bias,
                                            float* __restrict__ Of32,
                                            int wave, int lane) {
    const int nt  = wave & 3;
    const int mt0 = (wave >> 2) * 2, mt1 = mt0 + 1;
    const int nc  = lane & 15;
    const float bv = bias[nt * 16 + nc];
    const __bf16* bbase = Bpack + (size_t)nt * 8 * 512 + lane * 16;
    const __bf16* a0base = Abuf + (mt0 * 16 + (lane & 15)) * SH + ((lane >> 4) << 3);
    const __bf16* a1base = Abuf + (mt1 * 16 + (lane & 15)) * SH + ((lane >> 4) << 3);
    v8f acc0, acc1;
    #pragma unroll
    for (int e = 0; e < 8; ++e) { acc0[e] = bv; acc1[e] = bv; }
    #pragma unroll
    for (int kt = 0; kt < 8; ++kt) {
        v16bf b  = load_b_at(bbase + kt * 512);
        v16bf a0 = load_a_at(a0base + kt * 32);
        v16bf a1 = load_a_at(a1base + kt * 32);
        acc0 = __builtin_amdgcn_wmma_f32_16x16x32_bf16(false, a0, false, b,
                                                       (short)0, acc0, false, false);
        acc1 = __builtin_amdgcn_wmma_f32_16x16x32_bf16(false, a1, false, b,
                                                       (short)0, acc1, false, false);
    }
    const int r0 = mt0 * 16 + ((lane >> 4) << 3);
    const int r1 = mt1 * 16 + ((lane >> 4) << 3);
    #pragma unroll
    for (int e = 0; e < 8; ++e) {
        Of32[(r0 + e) * S3 + nt * 16 + nc] = acc0[e];
        Of32[(r1 + e) * S3 + nt * 16 + nc] = acc1[e];
    }
}

// ---------------- main persistent flow kernel ----------------
__global__ void __launch_bounds__(256, 1)
flow_kernel(const float* __restrict__ x,
            const float* __restrict__ logdet_in,
            const float* __restrict__ bvec,
            const float* __restrict__ logs,
            const float* __restrict__ b1,
            const float* __restrict__ b2,
            const float* __restrict__ bfv,
            const float* __restrict__ lgam,
            const __bf16* __restrict__ w1p,
            const __bf16* __restrict__ w2p,
            const __bf16* __restrict__ wfp,
            const float* __restrict__ sumlogs,
            float* __restrict__ out) {
    extern __shared__ char smem[];
    __bf16* A1  = (__bf16*)(smem + A1_OFF);
    __bf16* H1  = (__bf16*)(smem + H1_OFF);
    __bf16* H2  = (__bf16*)(smem + H2_OFF);
    float*  H3  = (float*) (smem + H1_OFF);   // alias over H1 (barrier-separated)
    float*  Z0p = (float*) (smem + Z0_OFF);
    float*  Z1p = (float*) (smem + Z1_OFF);

    const int t    = threadIdx.x;
    const int wave = t >> 5;
    const int lane = t & 31;
    const int r    = t >> 2;        // phase-E row (4 threads/row)
    const int c    = t & 3;         // 16-column chunk
    const int rowBase = blockIdx.x * 64;

    // ---- stage x tile into Z0 (vectorized) ----
    {
        const float4* xr = (const float4*)(x + (size_t)(rowBase + r) * LL + c * 16);
        float4* zp = (float4*)(Z0p + r * SZ + c * 16);
        #pragma unroll
        for (int q = 0; q < 4; ++q) zp[q] = xr[q];
    }
    float ld = 0.f;
    if (t < 64) ld = logdet_in[rowBase + t];
    __syncthreads();

    for (int l = 0; l < DD; ++l) {
        float* Zc = (l & 1) ? Z1p : Z0p;
        float* Zn = (l & 1) ? Z0p : Z1p;

        // ---- phase E: z = (z + b) * exp(logs); stage z1 bf16 into A1 ----
        {
            const float4* bl = (const float4*)(bvec + l * LL + c * 16);
            const float4* sl = (const float4*)(logs + l * LL + c * 16);
            float4* zp = (float4*)(Zc + r * SZ + c * 16);
            #pragma unroll
            for (int q = 0; q < 4; ++q) {
                float4 z = zp[q], bq = bl[q], sq = sl[q];
                z.x = (z.x + bq.x) * __expf(sq.x);
                z.y = (z.y + bq.y) * __expf(sq.y);
                z.z = (z.z + bq.z) * __expf(sq.z);
                z.w = (z.w + bq.w) * __expf(sq.w);
                zp[q] = z;
                if (c < 2) {   // z1 columns -> bf16 A-matrix
                    union { __bf16 h[4]; uint2 u; } pk;
                    pk.h[0] = (__bf16)z.x; pk.h[1] = (__bf16)z.y;
                    pk.h[2] = (__bf16)z.z; pk.h[3] = (__bf16)z.w;
                    *(uint2*)(A1 + r * SA + c * 16 + q * 4) = pk.u;
                }
            }
        }
        __syncthreads();

        // ---- GEMM1: (64x32)@(32x256)+b1, relu -> H1 ----
        gemm_nt2<1>(A1, SA, w1p + (size_t)l * 16 * 1 * 512, b1 + l * NW, H1, wave, lane);
        __syncthreads();

        // ---- GEMM2: (64x256)@(256x256)+b2, relu -> H2 ----
        gemm_nt2<8>(H1, SH, w2p + (size_t)l * 16 * 8 * 512, b2 + l * NW, H2, wave, lane);
        __syncthreads();

        // ---- GEMM3: (64x256)@(256x64)+bf -> H3 (f32, aliases H1) ----
        gemm3_stage(H2, wfp + (size_t)l * 4 * 8 * 512, bfv + l * 64, H3, wave, lane);
        __syncthreads();

        // ---- epilogue: coupling update, logdet, reversal-on-write ----
        if (t < 64) {
            const float* hrow = H3 + t * S3;
            const float* lg   = lgam + l * NS;
            const float* zc   = Zc + t * SZ;
            float*       zn   = Zn + t * SZ;
            ld += sumlogs[l];
            #pragma unroll
            for (int k = 0; k < 32; ++k) {
                float shift = __expf(lg[k]) * tanhf(hrow[2 * k]);
                float scale = ALPHA * tanhf(hrow[2 * k + 1]);
                float z2 = zc[32 + k];
                zn[31 - k] = z2 + scale * z2 + shift;   // reversed z2'
                ld += log1pf(scale);
            }
            #pragma unroll
            for (int j = 0; j < 32; ++j)
                zn[63 - j] = zc[j];                      // reversed z1
        }
        __syncthreads();
    }

    // ---- final store (z back in Z0 after 8 flips) ----
    {
        float4* orow = (float4*)(out + (size_t)(rowBase + r) * LL + c * 16);
        const float4* zp = (const float4*)(Z0p + r * SZ + c * 16);
        #pragma unroll
        for (int q = 0; q < 4; ++q) orow[q] = zp[q];
    }
    if (t < 64) out[(size_t)BB * LL + rowBase + t] = ld;
}

// ---------------- host launch ----------------
extern "C" void kernel_launch(void* const* d_in, const int* in_sizes, int n_in,
                              void* d_out, int out_size, void* d_ws, size_t ws_size,
                              hipStream_t stream) {
    const float* x      = (const float*)d_in[0];
    const float* logdet = (const float*)d_in[1];
    const float* bvec   = (const float*)d_in[2];
    const float* logs   = (const float*)d_in[3];
    const float* W1     = (const float*)d_in[4];
    const float* b1     = (const float*)d_in[5];
    const float* W2     = (const float*)d_in[6];
    const float* b2     = (const float*)d_in[7];
    const float* Wf     = (const float*)d_in[8];
    const float* bfv    = (const float*)d_in[9];
    const float* lgam   = (const float*)d_in[10];
    float* out = (float*)d_out;

    // workspace: packed bf16 weight fragments + sum(logs) table
    __bf16* w1p = (__bf16*)d_ws;                                   // 65536  bf16
    __bf16* w2p = (__bf16*)((char*)d_ws + 131072);                 // 524288 bf16
    __bf16* wfp = (__bf16*)((char*)d_ws + 131072 + 1048576);       // 131072 bf16
    float*  sl  = (float*)((char*)d_ws + 131072 + 1048576 + 262144);

    pack_w_kernel<<<(4096 + 255) / 256, 256, 0, stream>>>(W1, w1p, 32, 256, 4096);
    pack_w_kernel<<<(32768 + 255) / 256, 256, 0, stream>>>(W2, w2p, 256, 256, 32768);
    pack_w_kernel<<<(8192 + 255) / 256, 256, 0, stream>>>(Wf, wfp, 256, 64, 8192);
    sumlogs_kernel<<<1, DD, 0, stream>>>(logs, sl);

    flow_kernel<<<BB / 64, 256, SMEM_BYTES, stream>>>(
        x, logdet, bvec, logs, b1, b2, bfv, lgam, w1p, w2p, wfp, sl, out);
}
